// Pair_Interaction_Modify_25993142076014
// MI455X (gfx1250) — matrix operand order repaired and checked by reading, not compiled, not run on hardware
//
#include <hip/hip_runtime.h>
#include <hip/hip_bf16.h>
#include <stdint.h>

// ---------------- WMMA fragment types (gfx1250, wave32) ----------------
typedef __attribute__((ext_vector_type(16))) __bf16        bf16x16;
typedef __attribute__((ext_vector_type(2)))  __bf16        bf16x2;
typedef __attribute__((ext_vector_type(8)))  float         f32x8;
typedef __attribute__((ext_vector_type(4)))  unsigned int  u32x4;

union Frag {
    bf16x16 v;
    u32x4   q[2];   // 32 bytes = 8 VGPRs
};

__device__ __forceinline__ uint16_t f2bf(float f) {
    uint32_t u = __float_as_uint(f);
    return (uint16_t)((u + 0x7FFFu + ((u >> 16) & 1u)) >> 16);  // RNE
}
__device__ __forceinline__ float bf2f(uint16_t h) {
    return __uint_as_float(((uint32_t)h) << 16);
}
// packed f32x2 -> bf16x2 (single v_cvt_pk_bf16_f32 where available)
__device__ __forceinline__ uint32_t f2bf_pk(float lo, float hi) {
#if __has_builtin(__builtin_amdgcn_cvt_pk_bf16_f32)
    union { bf16x2 v; uint32_t u; } c;
    c.v = __builtin_amdgcn_cvt_pk_bf16_f32(lo, hi);
    return c.u;
#else
    return (uint32_t)f2bf(lo) | ((uint32_t)f2bf(hi) << 16);
#endif
}
// fast sigmoid: v_exp_f32 + v_rcp_f32 (no IEEE div_scale chain)
__device__ __forceinline__ float fast_sigmoid(float z) {
    return __builtin_amdgcn_rcpf(1.0f + __expf(-z));
}

#define DDIM  256
#define MT    32            // batch rows per workgroup
#define NR    192           // 6 pairs * MT rows
#define NRT   12            // row tiles of 16 in phase B
#define LROW  264           // bf16 elems per LDS row: 256 + 8 pad (528B, 16B aligned)

// ---------------- prep: f32 weights -> bf16 in workspace ----------------
__global__ __launch_bounds__(256) void convert_weights_kernel(
    const float* __restrict__ Wi, const float* __restrict__ Wf,
    const float* __restrict__ Wfc,
    uint16_t* __restrict__ WiB, uint16_t* __restrict__ WfB,
    uint16_t* __restrict__ WfcB)
{
    int i = blockIdx.x * 256 + threadIdx.x;
    if (i < 256 * 256) {
        WiB[i] = f2bf(Wi[i]);
        WfB[i] = f2bf(Wf[i]);
    }
    if (i < 256 * 1024) {
        WfcB[i] = f2bf(Wfc[i]);
    }
}

// ---------------- fused pair-interaction kernel ----------------
__global__ __launch_bounds__(256) void pair_interact_kernel(
    const float* __restrict__ x0, const float* __restrict__ x1,
    const float* __restrict__ x2, const float* __restrict__ x3,
    const float* __restrict__ bi, const float* __restrict__ bfv_,
    const float* __restrict__ bfc,
    const uint16_t* __restrict__ WiB, const uint16_t* __restrict__ WfB,
    const uint16_t* __restrict__ WfcB,
    float* __restrict__ out)
{
    __shared__ uint16_t Ssm[NR * LROW];   // pairwise sums  (bf16)  ~99 KB
    __shared__ uint16_t Usm[NR * LROW];   // tempt1         (bf16)  ~99 KB
    __shared__ float    nrm[NR];
    __shared__ int      sel[MT][4];

    const int tid   = threadIdx.x;
    const int lane  = tid & 31;
    const int wave  = tid >> 5;
    const int L     = lane & 15;   // N / row index within fragment
    const int hi    = lane >> 4;   // half-wave select
    const int bbase = blockIdx.x * MT;

    const float* xs[4] = {x0, x1, x2, x3};
    const int PI[6] = {0, 0, 0, 1, 1, 2};
    const int PJ[6] = {1, 2, 3, 2, 3, 3};

    // ---- Phase A: S[p*MT+m][d] = x_i[b][d] + x_j[b][d], stored bf16 ----
    #pragma unroll
    for (int p = 0; p < 6; ++p) {
        const float* xi = xs[PI[p]];
        const float* xj = xs[PJ[p]];
        for (int u = tid; u < MT * (DDIM / 2); u += 256) {
            int m = u >> 7;            // row within tile (D/2 = 128 float2 cols)
            int c = u & 127;
            float2 a = ((const float2*)(xi + (size_t)(bbase + m) * DDIM))[c];
            float2 b = ((const float2*)(xj + (size_t)(bbase + m) * DDIM))[c];
            ((uint32_t*)(Ssm + (p * MT + m) * LROW))[c] =
                f2bf_pk(a.x + b.x, a.y + b.y);
        }
    }
    __syncthreads();

    // ---- Phase B: gate/tmp GEMMs + elementwise -> U in LDS ----
    // Each wave owns 2 column tiles; weight fragments for the full K=256
    // (8 k-steps x {Wf, Wi} = 128 VGPRs) are loaded ONCE per column tile and
    // reused across all 12 row tiles -> Wi/Wf are read exactly once per WG.
    #pragma unroll
    for (int cc = 0; cc < 2; ++cc) {
        const int ct   = wave + 8 * cc;          // column tile 0..15
        const int ecol = ct * 16 + L;            // output feature / weight row
        Frag wg[8], wl[8];
        #pragma unroll
        for (int ks = 0; ks < 8; ++ks) {
            int w0 = ks * 32 + hi * 16;          // B layout: contiguous 16 K elems per lane
            wg[ks].q[0] = *(const u32x4*)(WfB + (size_t)ecol * DDIM + w0);
            wg[ks].q[1] = *(const u32x4*)(WfB + (size_t)ecol * DDIM + w0 + 8);
            wl[ks].q[0] = *(const u32x4*)(WiB + (size_t)ecol * DDIM + w0);
            wl[ks].q[1] = *(const u32x4*)(WiB + (size_t)ecol * DDIM + w0 + 8);
        }
        const float bgate = bfv_[ecol];
        const float blin  = bi[ecol];

        for (int rt = 0; rt < NRT; ++rt) {
            f32x8 accg = {0.f,0.f,0.f,0.f,0.f,0.f,0.f,0.f};
            f32x8 acct = {0.f,0.f,0.f,0.f,0.f,0.f,0.f,0.f};
            const uint16_t* arow = Ssm + (rt * 16 + L) * LROW;  // A: row M=L
            #pragma unroll
            for (int ks = 0; ks < 8; ++ks) {
                Frag a;
                int d0 = ks * 32 + hi * 16;  // A layout: contiguous 16 K elems per lane
                a.q[0] = *(const u32x4*)(arow + d0);
                a.q[1] = *(const u32x4*)(arow + d0 + 8);
                accg = __builtin_amdgcn_wmma_f32_16x16x32_bf16(
                           false, a.v, false, wg[ks].v, (short)0, accg, false, false);
                acct = __builtin_amdgcn_wmma_f32_16x16x32_bf16(
                           false, a.v, false, wl[ks].v, (short)0, acct, false, false);
            }
            int r0 = rt * 16 + hi * 8;       // C/D layout: M = v + 8*hi, N = L
            #pragma unroll
            for (int v = 0; v < 8; ++v) {
                int r = r0 + v;
                float g  = fast_sigmoid(accg[v] + bgate);
                float tm = acct[v] + blin;
                float sv = bf2f(Ssm[r * LROW + ecol]);
                float uu = fmaxf(g * tm + (1.0f - g) * sv, 0.0f);
                Usm[r * LROW + ecol] = f2bf(uu);
            }
        }
    }
    __syncthreads();

    // ---- Norms: sum of squares per row (sqrt not needed for ranking) ----
    if (tid < NR) {
        const uint32_t* pp = (const uint32_t*)(Usm + tid * LROW);
        float s = 0.f;
        for (int k = 0; k < DDIM / 2; ++k) {
            uint32_t w = pp[k];
            float f0 = bf2f((uint16_t)(w & 0xFFFFu));
            float f1 = bf2f((uint16_t)(w >> 16));
            s = fmaf(f0, f0, s);
            s = fmaf(f1, f1, s);
        }
        nrm[tid] = s;
    }
    __syncthreads();

    // ---- Top-4 of 6 per batch row, descending, stable (matches top_k) ----
    if (tid < MT) {
        float vv[6];
        bool  used[6];
        #pragma unroll
        for (int p = 0; p < 6; ++p) { vv[p] = nrm[p * MT + tid]; used[p] = false; }
        #pragma unroll
        for (int q = 0; q < 4; ++q) {
            int best = 0; float bv = -1.0f;
            #pragma unroll
            for (int p = 0; p < 6; ++p)
                if (!used[p] && vv[p] > bv) { bv = vv[p]; best = p; }
            used[best] = true;
            sel[tid][q] = best;
        }
    }
    __syncthreads();

    // ---- Phase C: out = concat(top4 U rows) @ Wfc^T + bfc ----
    // 2 row tiles x 16 col tiles = 32 tiles -> 4 per wave, K = 1024.
    // Wfc is read twice per WG (once per row tile).
    for (int t = wave; t < 32; t += 8) {
        const int rt2  = t >> 4;             // row tile 0/1 (batch rows)
        const int ct   = t & 15;
        f32x8 acc = {0.f,0.f,0.f,0.f,0.f,0.f,0.f,0.f};
        const int ocol = ct * 16 + L;
        const int ml   = rt2 * 16 + L;       // batch row within tile for this lane
        const uint16_t* wrow = WfcB + (size_t)ocol * 1024;
        #pragma unroll 4
        for (int kk = 0; kk < 1024; kk += 32) {
            int q  = kk >> 8;                // which of the 4 selected rows
            int kr = kk & 255;               // offset within that row
            int p  = sel[ml][q];             // per-lane gather of the selected pair
            const uint16_t* arow = Usm + (p * MT + ml) * LROW;
            Frag a, b;
            int d0 = kr + hi * 16;
            a.q[0] = *(const u32x4*)(arow + d0);
            a.q[1] = *(const u32x4*)(arow + d0 + 8);
            int w0 = kk + hi * 16;
            b.q[0] = *(const u32x4*)(wrow + w0);
            b.q[1] = *(const u32x4*)(wrow + w0 + 8);
            acc = __builtin_amdgcn_wmma_f32_16x16x32_bf16(
                      false, a.v, false, b.v, (short)0, acc, false, false);
        }
        float bias = bfc[ocol];
        #pragma unroll
        for (int v = 0; v < 8; ++v) {
            int m = rt2 * 16 + v + hi * 8;
            out[(size_t)(bbase + m) * 256 + ocol] = acc[v] + bias;
        }
    }
}

// ---------------- launch ----------------
extern "C" void kernel_launch(void* const* d_in, const int* in_sizes, int n_in,
                              void* d_out, int out_size, void* d_ws, size_t ws_size,
                              hipStream_t stream) {
    (void)in_sizes; (void)n_in; (void)out_size; (void)ws_size;
    const float* x0  = (const float*)d_in[0];
    const float* x1  = (const float*)d_in[1];
    const float* x2  = (const float*)d_in[2];
    const float* x3  = (const float*)d_in[3];
    const float* Wi  = (const float*)d_in[4];
    const float* bi  = (const float*)d_in[5];
    const float* Wf  = (const float*)d_in[6];
    const float* bf  = (const float*)d_in[7];
    const float* Wfc = (const float*)d_in[8];
    const float* bfc = (const float*)d_in[9];

    uint16_t* WiB  = (uint16_t*)d_ws;            // 256*256 bf16
    uint16_t* WfB  = WiB + 256 * 256;            // 256*256 bf16
    uint16_t* WfcB = WfB + 256 * 256;            // 256*1024 bf16  (total 768 KB)

    convert_weights_kernel<<<1024, 256, 0, stream>>>(Wi, Wf, Wfc, WiB, WfB, WfcB);

    const int B = 65536;
    pair_interact_kernel<<<B / MT, 256, 0, stream>>>(
        x0, x1, x2, x3, bi, bf, bfc, WiB, WfB, WfcB, (float*)d_out);
}
